// ScaledSTEQuantLinear_14078902796908
// MI455X (gfx1250) — 8-bit matrix operand order repaired and checked by reading, compiled, not run on hardware
//
#include <hip/hip_runtime.h>


typedef __attribute__((ext_vector_type(8))) int v8i_t;

#ifndef __has_builtin
#define __has_builtin(x) 0
#endif
#if __has_builtin(__builtin_amdgcn_global_load_async_to_lds_b128)
#define HAVE_ASYNC_LDS 1
#else
#define HAVE_ASYNC_LDS 0
#endif

constexpr int D_IN    = 4096;
constexpr int D_OUT   = 4096;
constexpr int M_ROWS  = 8192;
constexpr int QG      = 128;
constexpr int NGROUPS = D_IN / QG;

#if HAVE_ASYNC_LDS
typedef int v4si __attribute__((vector_size(16)));
typedef __attribute__((address_space(1))) v4si gv4_t;
typedef __attribute__((address_space(3))) v4si lv4_t;

__device__ __forceinline__ void async_cp16(const void* g, void* l) {
    __builtin_amdgcn_global_load_async_to_lds_b128(
        (gv4_t*)(unsigned long long)(size_t)g,
        (lv4_t*)(unsigned)(size_t)l, 0, 0);
}
__device__ __forceinline__ void wait_async0() {
#if __has_builtin(__builtin_amdgcn_s_wait_asynccnt)
    __builtin_amdgcn_s_wait_asynccnt(0);
#else
    asm volatile("s_wait_asynccnt 0x0" ::: "memory");
#endif
}
#endif

__global__ void scale_kernel(const float* __restrict__ log_scale,
                             float* __restrict__ scale,
                             float* __restrict__ inv_scale) {
    int k = blockIdx.x * blockDim.x + threadIdx.x;
    float s = expf(log_scale[k]);
    s = fminf(fmaxf(s, 1e-4f), 1e4f);
    scale[k]     = s;
    inv_scale[k] = 1.0f / s;
}

__global__ __launch_bounds__(256)
void act_quant_kernel(const float* __restrict__ x,
                      const float* __restrict__ inv_scale,
                      signed char* __restrict__ qx,
                      float* __restrict__ sx) {
    const int m = blockIdx.x;
    const int t = threadIdx.x;
    const float* xr = x + (size_t)m * D_IN;

    float v[16];
    float amax = 0.0f;
#pragma unroll
    for (int i = 0; i < 4; ++i) {
        int k = (t + i * 256) * 4;
        float4 xv = *(const float4*)(xr + k);
        float4 iv = *(const float4*)(inv_scale + k);
        v[i * 4 + 0] = xv.x * iv.x;
        v[i * 4 + 1] = xv.y * iv.y;
        v[i * 4 + 2] = xv.z * iv.z;
        v[i * 4 + 3] = xv.w * iv.w;
#pragma unroll
        for (int j = 0; j < 4; ++j) amax = fmaxf(amax, fabsf(v[i * 4 + j]));
    }
#pragma unroll
    for (int off = 16; off >= 1; off >>= 1)
        amax = fmaxf(amax, __shfl_xor(amax, off, 32));
    __shared__ float wmax[8];
    if ((t & 31) == 0) wmax[t >> 5] = amax;
    __syncthreads();
    float rmax = 1e-5f;
#pragma unroll
    for (int w = 0; w < 8; ++w) rmax = fmaxf(rmax, wmax[w]);

    const float s     = rmax * (1.0f / 127.0f);
    const float inv_s = 127.0f / rmax;
    if (t == 0) sx[m] = s;

    unsigned int* qrow = (unsigned int*)(qx + (size_t)m * D_IN);
#pragma unroll
    for (int i = 0; i < 4; ++i) {
        unsigned int packed = 0;
#pragma unroll
        for (int j = 0; j < 4; ++j) {
            int q = (int)rintf(v[i * 4 + j] * inv_s);
            q = q < -128 ? -128 : (q > 127 ? 127 : q);
            packed |= ((unsigned int)(q & 0xFF)) << (8 * j);
        }
        qrow[t + i * 256] = packed;
    }
}

__global__ __launch_bounds__(256)
void wt_quant_kernel(const float* __restrict__ weight,
                     const float* __restrict__ scale,
                     signed char* __restrict__ qw,
                     float* __restrict__ sw) {
    const int wid  = threadIdx.x >> 5;
    const int lid  = threadIdx.x & 31;
    const int gidx = blockIdx.x * 8 + wid;
    const int n    = gidx >> 5;
    const int g    = gidx & 31;
    const int kb   = g * QG + lid * 4;

    float4 wv = *(const float4*)(weight + (size_t)n * D_IN + kb);
    float4 sv = *(const float4*)(scale + kb);
    float w[4] = { wv.x * sv.x, wv.y * sv.y, wv.z * sv.z, wv.w * sv.w };

    float mx = fmaxf(fmaxf(w[0], w[1]), fmaxf(w[2], w[3]));
    float mn = fminf(fminf(w[0], w[1]), fminf(w[2], w[3]));
#pragma unroll
    for (int off = 16; off >= 1; off >>= 1) {
        mx = fmaxf(mx, __shfl_xor(mx, off, 32));
        mn = fminf(mn, __shfl_xor(mn, off, 32));
    }
    const float s = fmaxf(mx - mn, 1e-5f) * (1.0f / 15.0f);
    const float z = fminf(fmaxf(-rintf(mn / s), 0.0f), 15.0f);

    unsigned int packed = 0;
#pragma unroll
    for (int j = 0; j < 4; ++j) {
        float qf = fminf(fmaxf(rintf(w[j] / s) + z, 0.0f), 15.0f) - z;
        int   q  = (int)qf;
        packed |= ((unsigned int)(q & 0xFF)) << (8 * j);
    }
    ((unsigned int*)(qw + (size_t)n * D_IN))[g * 32 + lid] = packed;
    if (lid == 0) sw[g * D_OUT + n] = s;
}

#define BM 128
#define BN 128

__global__ __launch_bounds__(256)
void gemm_i8_kernel(const signed char* __restrict__ qx,
                    const signed char* __restrict__ qw,
                    const float* __restrict__ sx,
                    const float* __restrict__ sw,
                    const float* __restrict__ bias,
                    float* __restrict__ out) {
    constexpr int LDA = QG + 16;
    __shared__ __attribute__((aligned(16))) signed char ldsA[2][BM * LDA];
    __shared__ __attribute__((aligned(16))) signed char ldsB[2][BN * LDA];

    const int t     = threadIdx.x;
    const int lane  = t & 31;
    const int wid   = t >> 5;
    const int waveM = wid & 1;
    const int waveN = wid >> 1;
    const int half  = lane >> 4;
    const int l16   = lane & 15;

    const int blockM = blockIdx.y * BM;
    const int blockN = blockIdx.x * BN;

    float facc[8][8];
#pragma unroll
    for (int i = 0; i < 8; ++i)
#pragma unroll
        for (int r = 0; r < 8; ++r) facc[i][r] = 0.0f;

    const int row0 = t >> 3, seg0 = (t & 7) * 16;

#if HAVE_ASYNC_LDS
    auto issueTiles = [&](int buf, int kg) {
#pragma unroll
        for (int i = 0; i < 4; ++i) {
            int row = row0 + i * 32;
            async_cp16(qx + (size_t)(blockM + row) * D_IN + kg * QG + seg0,
                       &ldsA[buf][row * LDA + seg0]);
            async_cp16(qw + (size_t)(blockN + row) * D_IN + kg * QG + seg0,
                       &ldsB[buf][row * LDA + seg0]);
        }
    };
#else
    uint4 regA[4], regB[4];
    auto loadTiles = [&](int kg) {
#pragma unroll
        for (int i = 0; i < 4; ++i) {
            int row = row0 + i * 32;
            regA[i] = *(const uint4*)(qx + (size_t)(blockM + row) * D_IN + kg * QG + seg0);
            regB[i] = *(const uint4*)(qw + (size_t)(blockN + row) * D_IN + kg * QG + seg0);
        }
    };
    auto stageTiles = [&](int buf) {
#pragma unroll
        for (int i = 0; i < 4; ++i) {
            int row = row0 + i * 32;
            *(uint4*)(&ldsA[buf][row * LDA + seg0]) = regA[i];
            *(uint4*)(&ldsB[buf][row * LDA + seg0]) = regB[i];
        }
    };
#endif

#if HAVE_ASYNC_LDS
    issueTiles(0, 0);
#else
    loadTiles(0);
#endif
    int buf = 0;

    for (int kg = 0; kg < NGROUPS; ++kg) {
#if HAVE_ASYNC_LDS
        wait_async0();
        __syncthreads();
        if (kg + 1 < NGROUPS) issueTiles(buf ^ 1, kg + 1);
#else
        stageTiles(buf);
        __syncthreads();
        if (kg + 1 < NGROUPS) loadTiles(kg + 1);
#endif

        float swv[2];
#pragma unroll
        for (int tn = 0; tn < 2; ++tn)
            swv[tn] = sw[kg * D_OUT + (blockN + waveN * 32 + tn * 16 + l16)];

        const signed char* A = &ldsA[buf][0];
        const signed char* B = &ldsB[buf][0];

        v8i_t iacc[8];
#pragma unroll
        for (int i = 0; i < 8; ++i)
#pragma unroll
            for (int r = 0; r < 8; ++r) iacc[i][r] = 0;

#pragma unroll
        for (int ks = 0; ks < 2; ++ks) {
            v8i_t afrag[4];
#pragma unroll
            for (int tm = 0; tm < 4; ++tm) {
                const signed char* base = A + (waveM * 64 + tm * 16 + l16) * LDA + ks * 64;
                uint2 d0 = *(const uint2*)(base + 0  + 8 * half);
                uint2 d1 = *(const uint2*)(base + 16 + 8 * half);
                uint2 d2 = *(const uint2*)(base + 32 + 8 * half);
                uint2 d3 = *(const uint2*)(base + 48 + 8 * half);
                afrag[tm][0] = (int)d0.x; afrag[tm][1] = (int)d0.y;
                afrag[tm][2] = (int)d1.x; afrag[tm][3] = (int)d1.y;
                afrag[tm][4] = (int)d2.x; afrag[tm][5] = (int)d2.y;
                afrag[tm][6] = (int)d3.x; afrag[tm][7] = (int)d3.y;
            }
            v8i_t bfrag[2];
#pragma unroll
            for (int tn = 0; tn < 2; ++tn) {
                const signed char* base = B + (waveN * 32 + tn * 16 + l16) * LDA + ks * 64;
                uint2 c0 = *(const uint2*)(base + 0  + 8 * half);
                uint2 c1 = *(const uint2*)(base + 16 + 8 * half);
                uint2 c2 = *(const uint2*)(base + 32 + 8 * half);
                uint2 c3 = *(const uint2*)(base + 48 + 8 * half);
                bfrag[tn][0] = (int)c0.x; bfrag[tn][1] = (int)c0.y;
                bfrag[tn][2] = (int)c1.x; bfrag[tn][3] = (int)c1.y;
                bfrag[tn][4] = (int)c2.x; bfrag[tn][5] = (int)c2.y;
                bfrag[tn][6] = (int)c3.x; bfrag[tn][7] = (int)c3.y;
            }
#pragma unroll
            for (int tm = 0; tm < 4; ++tm)
#pragma unroll
                for (int tn = 0; tn < 2; ++tn)
                    iacc[tm * 2 + tn] = __builtin_amdgcn_wmma_i32_16x16x64_iu8(
                        true, afrag[tm], true, bfrag[tn], iacc[tm * 2 + tn],
                        false, false);
        }

#pragma unroll
        for (int tm = 0; tm < 4; ++tm)
#pragma unroll
            for (int tn = 0; tn < 2; ++tn)
#pragma unroll
                for (int r = 0; r < 8; ++r)
                    facc[tm * 2 + tn][r] += (float)iacc[tm * 2 + tn][r] * swv[tn];

#if !HAVE_ASYNC_LDS
        __syncthreads();
#endif
        buf ^= 1;
    }

#pragma unroll
    for (int tm = 0; tm < 4; ++tm) {
        const int m0 = blockM + waveM * 64 + tm * 16 + 8 * half;
#pragma unroll
        for (int tn = 0; tn < 2; ++tn) {
            const int n = blockN + waveN * 32 + tn * 16 + l16;
            const float bv = bias[n];
#pragma unroll
            for (int r = 0; r < 8; ++r) {
                const int m = m0 + r;
                out[(size_t)m * D_OUT + n] = sx[m] * facc[tm * 2 + tn][r] + bv;
            }
        }
    }
}

extern "C" void kernel_launch(void* const* d_in, const int* in_sizes, int n_in,
                              void* d_out, int out_size, void* d_ws, size_t ws_size,
                              hipStream_t stream) {
    const float* x         = (const float*)d_in[0];
    const float* weight    = (const float*)d_in[1];
    const float* bias      = (const float*)d_in[2];
    const float* log_scale = (const float*)d_in[3];
    float*       out       = (float*)d_out;

    char* ws = (char*)d_ws;
    float* scale     = (float*)ws;  ws += (size_t)D_IN * sizeof(float);
    float* inv_scale = (float*)ws;  ws += (size_t)D_IN * sizeof(float);
    float* sx        = (float*)ws;  ws += (size_t)M_ROWS * sizeof(float);
    float* sw        = (float*)ws;  ws += (size_t)D_OUT * NGROUPS * sizeof(float);
    signed char* qx  = (signed char*)ws;  ws += (size_t)M_ROWS * D_IN;
    signed char* qw  = (signed char*)ws;

    scale_kernel<<<D_IN / 256, 256, 0, stream>>>(log_scale, scale, inv_scale);
    act_quant_kernel<<<M_ROWS, 256, 0, stream>>>(x, inv_scale, qx, sx);
    wt_quant_kernel<<<(D_OUT * NGROUPS) / 8, 256, 0, stream>>>(weight, scale, qw, sw);
    gemm_i8_kernel<<<dim3(D_OUT / BN, M_ROWS / BM), 256, 0, stream>>>(qx, qw, sx, sw, bias, out);
}
